// AttentiveRelativeRole_35029753266257
// MI455X (gfx1250) — compile-verified, matrix operand order checked by reading, not run on hardware
//
#include <hip/hip_runtime.h>
#include <hip/hip_bf16.h>

typedef __bf16 bf16_t;
typedef __attribute__((ext_vector_type(16))) __bf16 v16bf;
typedef __attribute__((ext_vector_type(8)))  __bf16 v8bf;
typedef __attribute__((ext_vector_type(8)))  float  v8f;
typedef __attribute__((ext_vector_type(4)))  unsigned int v4u;
typedef __attribute__((ext_vector_type(8)))  int v8i;
typedef __attribute__((ext_vector_type(4)))  int v4i;

#define B_   4
#define S_   1024
#define H_   16
#define DK_  64
#define V_   2048
#define EPS_ 1e-6f

// ---------------------------------------------------------------------------
// WMMA fragment helpers (CDNA5 wave32 16-bit A/B layout):
//   lane 0-15 :  M(or N)=lane,     elements 0..7 -> K 0..7,   8..15 -> K 16..23
//   lane 16-31:  M(or N)=lane-16,  elements 0..7 -> K 8..15,  8..15 -> K 24..31
// i.e. two contiguous 16-byte chunks at k-offsets koff and koff+16,
// koff = 8*(lane>=16). Caller passes pointer to [row][k_base + koff].
// ---------------------------------------------------------------------------
__device__ __forceinline__ v16bf make_frag(const bf16_t* p) {
  v8bf lo = *(const v8bf*)p;
  v8bf hi = *(const v8bf*)(p + 16);
  v16bf r;
#pragma unroll
  for (int i = 0; i < 8; ++i) { r[i] = lo[i]; r[i + 8] = hi[i]; }
  return r;
}

__device__ __forceinline__ v8f wmma_bf16(v16bf a, v16bf b, v8f c) {
  return __builtin_amdgcn_wmma_f32_16x16x32_bf16(false, a, false, b,
                                                 (short)0, c, false, false);
}

// ---------------------------------------------------------------------------
// TDM: issue a 2D TENSOR_LOAD_TO_LDS of a [rows x 32] bf16 tile whose rows are
// `stride` elements apart in memory, into LDS at `lds_byte_off`, padding each
// 64-byte row with 16 bytes (-> LDS row pitch 40 bf16, matching As/Wt tiles).
// Descriptor bit layout per CDNA5 ISA 8.3/8.4 (D# group 0/1); groups 2/3 zero.
// Wave-level op (EXEC ignored); call from one wave, tracked by TENSORcnt.
// ---------------------------------------------------------------------------
__device__ __forceinline__ void tdm_load_tile(const bf16_t* gsrc, unsigned lds_byte_off,
                                              int rows, int stride) {
  unsigned long long ga = (unsigned long long)(uintptr_t)gsrc;   // byte address
  v4u g0 = { 1u,                                   // count=1, user descriptor
             lds_byte_off,                         // lds_addr
             (unsigned)(ga & 0xFFFFFFFFu),         // global_addr[31:0]
             (unsigned)((ga >> 32) & 0x01FFFFFFu) | (2u << 30) }; // addr[56:32], type=2
  v8i g1;
  g1[0] = (int)((1u << 16)        // data_size = 1 -> 2 bytes
              | (1u << 20)        // pad_enable
              | (3u << 22)        // pad_interval: 2^(3+1)=16 dwords = 64B
              | (3u << 25));      // pad_amount: 4 dwords = 16B (8 bf16)
  g1[1] = (int)(((unsigned)stride & 0xFFFFu) << 16);          // tensor_dim0[15:0]
  g1[2] = (int)(((unsigned)stride >> 16) |
                (((unsigned)rows & 0xFFFFu) << 16));          // td0[31:16] | td1[15:0]
  g1[3] = (int)(32u << 16);                                   // tile_dim0 = 32
  g1[4] = (int)((unsigned)rows & 0xFFFFu);                    // tile_dim1 = rows
  g1[5] = (int)stride;                                        // dim0_stride[31:0]
  g1[6] = 0;                                                  // stride hi / dim1_stride
  g1[7] = 0;
  v4i z4 = {0, 0, 0, 0};
#if __clang_major__ >= 23
  v8i z8 = {0, 0, 0, 0, 0, 0, 0, 0};
  __builtin_amdgcn_tensor_load_to_lds(g0, g1, z4, z4, z8, 0);
#else
  __builtin_amdgcn_tensor_load_to_lds(g0, g1, z4, z4, 0);
#endif
}

// ---------------------------------------------------------------------------
// Elementwise prep: (x + pos) -> bf16 ; weights -> bf16
// ---------------------------------------------------------------------------
__global__ __launch_bounds__(256) void addcvt_kernel(const float* __restrict__ x,
                                                     const float* __restrict__ pos,
                                                     bf16_t* __restrict__ out, int n) {
  int i = blockIdx.x * 256 + threadIdx.x;
  if (i < n) out[i] = (bf16_t)(x[i] + pos[i]);
}

__global__ __launch_bounds__(256) void cvt_kernel(const float* __restrict__ x,
                                                  bf16_t* __restrict__ out, int n) {
  int i = blockIdx.x * 256 + threadIdx.x;
  if (i < n) out[i] = (bf16_t)x[i];
}

// ---------------------------------------------------------------------------
// Codebook prep: cbn[v] = l2norm(cb[v]) (bf16)   and
//                F[v]   = LN( LN(cb[v], vq_g, vq_b) @ Wo , ln_g, ln_b )
// (forward STE output y == codebook[id], so quantized depends only on id)
// ---------------------------------------------------------------------------
__global__ __launch_bounds__(64) void cb_prep(const float* __restrict__ cb,
                                              const float* __restrict__ vqg,
                                              const float* __restrict__ vqb,
                                              const float* __restrict__ Wo,
                                              const float* __restrict__ lng,
                                              const float* __restrict__ lnb,
                                              bf16_t* __restrict__ cbn,
                                              float* __restrict__ Ftab) {
  __shared__ float row[64], lnr[64], emb[64];
  int v = blockIdx.x, j = threadIdx.x;
  row[j] = cb[(size_t)v * DK_ + j];
  __syncthreads();
  float m = 0.f, sq = 0.f;
  for (int i = 0; i < 64; ++i) { float x = row[i]; m += x; sq += x * x; }
  m *= (1.f / 64.f);
  float var = sq * (1.f / 64.f) - m * m;
  float x = row[j];
  cbn[(size_t)v * DK_ + j] = (bf16_t)(x * rsqrtf(fmaxf(sq, 1e-12f)));
  lnr[j] = (x - m) * rsqrtf(var + EPS_) * vqg[j] + vqb[j];
  __syncthreads();
  float e = 0.f;
  for (int i = 0; i < 64; ++i) e += lnr[i] * Wo[i * 64 + j];
  emb[j] = e;
  __syncthreads();
  float m2 = 0.f, sq2 = 0.f;
  for (int i = 0; i < 64; ++i) { float y = emb[i]; m2 += y; sq2 += y * y; }
  m2 *= (1.f / 64.f);
  float v2 = sq2 * (1.f / 64.f) - m2 * m2;
  Ftab[(size_t)v * DK_ + j] = (e - m2) * rsqrtf(v2 + EPS_) * lng[j] + lnb[j];
}

// ---------------------------------------------------------------------------
// Projection GEMM: C[M,N] = A[M,Kd] @ W[Kd,N], bf16 in, f32 acc, bf16 out.
// 256 threads (8 waves), 128x128 tile, KT=32. Wave = 64(M) x 32(N) outputs.
// Double-buffered: next A tile via TDM (wave 0, TENSORcnt), next W tile via
// manual transpose staging, both overlapped with WMMAs on the current tile.
// mode 0: out[b][h][s][dk]   (q, k)      mode 1: out[b][h][dk][s]  (v^T)
// ---------------------------------------------------------------------------
__global__ __launch_bounds__(256) void proj_gemm(const bf16_t* __restrict__ A,
                                                 const bf16_t* __restrict__ W,
                                                 bf16_t* __restrict__ out,
                                                 int M, int Kd, int N, int S, int mode) {
  __shared__ alignas(16) bf16_t As[2][128][40];  // 128 rows x 32 k (+8 pad via TDM)
  __shared__ alignas(16) bf16_t Wt[2][128][40];  // 128 n    x 32 k (transposed W)
  int t = threadIdx.x;
  int lane = t & 31, w = t >> 5;
  int m0 = blockIdx.x * 128, n0 = blockIdx.y * 128;
  int m_off = (w & 1) * 64, n_off = (w >> 1) * 32;
  int mrow = lane & 15, koff = (lane >> 4) << 3;

  v8f acc[4][2];
#pragma unroll
  for (int a = 0; a < 4; ++a)
#pragma unroll
    for (int c = 0; c < 2; ++c)
#pragma unroll
      for (int e = 0; e < 8; ++e) acc[a][c][e] = 0.f;

  int wk = t >> 3, wn = (t & 7) * 16;        // W staging: 32 k x 8 n-chunks

  auto stageW = [&](int k0, int buf) {
    const bf16_t* src = W + (size_t)(k0 + wk) * N + n0 + wn;
    v8bf w0 = *(const v8bf*)src;
    v8bf w1 = *(const v8bf*)(src + 8);
#pragma unroll
    for (int j = 0; j < 8; ++j) { Wt[buf][wn + j][wk] = w0[j]; Wt[buf][wn + 8 + j][wk] = w1[j]; }
  };

  const int NK = Kd / 32;
  // prologue: stage tile 0 into buffer 0
  if (w == 0)
    tdm_load_tile(A + (size_t)m0 * Kd, (unsigned)(uintptr_t)&As[0][0][0], 128, Kd);
  stageW(0, 0);
  __builtin_amdgcn_s_wait_tensorcnt(0);
  __syncthreads();

  for (int i = 0; i < NK; ++i) {
    int buf = i & 1;
    if (i + 1 < NK) {   // overlap next-tile staging with compute
      if (w == 0)
        tdm_load_tile(A + (size_t)m0 * Kd + (i + 1) * 32,
                      (unsigned)(uintptr_t)&As[buf ^ 1][0][0], 128, Kd);
      stageW((i + 1) * 32, buf ^ 1);
      __builtin_prefetch(W + (size_t)((i + 2 < NK ? (i + 2) * 32 : 0) + wk) * N + n0 + wn, 0, 1);
    }
    v16bf af[4], wf[2];
#pragma unroll
    for (int mi = 0; mi < 4; ++mi) af[mi] = make_frag(&As[buf][m_off + mi * 16 + mrow][koff]);
#pragma unroll
    for (int ni = 0; ni < 2; ++ni) wf[ni] = make_frag(&Wt[buf][n_off + ni * 16 + mrow][koff]);
#pragma unroll
    for (int mi = 0; mi < 4; ++mi)
#pragma unroll
      for (int ni = 0; ni < 2; ++ni)
        acc[mi][ni] = wmma_bf16(af[mi], wf[ni], acc[mi][ni]);
    __builtin_amdgcn_s_wait_tensorcnt(0);   // next A tile landed (wave 0; no-op others)
    __syncthreads();                        // hand over buffers
  }
  // store (C layout: lane l, vgpr r -> row = r + 8*(l>=16), col = l&15)
#pragma unroll
  for (int mi = 0; mi < 4; ++mi) {
#pragma unroll
    for (int ni = 0; ni < 2; ++ni) {
      int n = n0 + n_off + ni * 16 + mrow;
      int h = n >> 6, dk = n & 63;
#pragma unroll
      for (int r = 0; r < 8; ++r) {
        int m = m0 + m_off + mi * 16 + r + koff;
        int bb = m / S, s = m - bb * S;
        size_t off = (mode == 0)
            ? ((((size_t)bb * H_ + h) * S + s) * DK_ + dk)
            : ((((size_t)bb * H_ + h) * DK_ + dk) * (size_t)S + s);
        out[off] = (bf16_t)acc[mi][ni][r];
      }
    }
  }
}

// ---------------------------------------------------------------------------
// Flash attention per (b,h): role = softmax(Q K^T + mask) V
// WG = 128 threads (4 waves), each wave owns 16 q rows; K-blocks of 128.
// Q [bh][s][dk], K [bh][s][dk], Vt [bh][dk][s]   (all bf16, contiguous K-dim
// for direct-from-global WMMA fragments). Outputs role (bf16) + row rsqrt-norm.
// ---------------------------------------------------------------------------
__global__ __launch_bounds__(128) void attn_kernel(const bf16_t* __restrict__ Qh,
                                                   const bf16_t* __restrict__ Kh,
                                                   const bf16_t* __restrict__ Vt,
                                                   const float* __restrict__ mask,
                                                   bf16_t* __restrict__ roleb,
                                                   float* __restrict__ rsq) {
  __shared__ alignas(16) bf16_t Ps[4][16][136];  // per-wave P tile (D->A relayout)
  int lane = threadIdx.x & 31, w = threadIdx.x >> 5;
  int bh = blockIdx.y;
  int bb = bh >> 4;  // batch (H=16)
  const bf16_t* qp = Qh + (size_t)bh * S_ * DK_;
  const bf16_t* kp = Kh + (size_t)bh * S_ * DK_;
  const bf16_t* vp = Vt + (size_t)bh * DK_ * S_;
  const float*  mp = mask + (size_t)bb * S_ * S_;
  int q0 = blockIdx.x * 64 + w * 16;
  int mrow = lane & 15, koff = (lane >> 4) << 3;

  v16bf qa0 = make_frag(qp + (size_t)(q0 + mrow) * DK_ + koff);
  v16bf qa1 = make_frag(qp + (size_t)(q0 + mrow) * DK_ + 32 + koff);

  float mr[8], lr[8];
  v8f o[4];
#pragma unroll
  for (int r = 0; r < 8; ++r) { mr[r] = -__builtin_inff(); lr[r] = 0.f; }
#pragma unroll
  for (int dt = 0; dt < 4; ++dt)
#pragma unroll
    for (int e = 0; e < 8; ++e) o[dt][e] = 0.f;

  for (int kb = 0; kb < S_; kb += 128) {
    v8f s[8];
#pragma unroll
    for (int nt = 0; nt < 8; ++nt) {
#pragma unroll
      for (int e = 0; e < 8; ++e) s[nt][e] = 0.f;
      const bf16_t* kr = kp + (size_t)(kb + nt * 16 + mrow) * DK_;
      s[nt] = wmma_bf16(qa0, make_frag(kr + koff), s[nt]);
      s[nt] = wmma_bf16(qa1, make_frag(kr + 32 + koff), s[nt]);
    }
    // + mask (faithful; mask broadcast over heads)
#pragma unroll
    for (int r = 0; r < 8; ++r) {
      const float* mrp = mp + (size_t)(q0 + r + koff) * S_ + kb;
#pragma unroll
      for (int nt = 0; nt < 8; ++nt) s[nt][r] += mrp[nt * 16 + mrow];
    }
    // online softmax update (row spread over 16-lane half; xor-shuffle reduce)
    float sc[8];
#pragma unroll
    for (int r = 0; r < 8; ++r) {
      float mx = s[0][r];
#pragma unroll
      for (int nt = 1; nt < 8; ++nt) mx = fmaxf(mx, s[nt][r]);
#pragma unroll
      for (int off = 1; off < 16; off <<= 1) mx = fmaxf(mx, __shfl_xor(mx, off, 32));
      float mn = fmaxf(mr[r], mx);
      sc[r] = __expf(mr[r] - mn);
      float rs = 0.f;
#pragma unroll
      for (int nt = 0; nt < 8; ++nt) { float pv = __expf(s[nt][r] - mn); s[nt][r] = pv; rs += pv; }
#pragma unroll
      for (int off = 1; off < 16; off <<= 1) rs += __shfl_xor(rs, off, 32);
      lr[r] = lr[r] * sc[r] + rs;
      mr[r] = mn;
    }
#pragma unroll
    for (int dt = 0; dt < 4; ++dt)
#pragma unroll
      for (int r = 0; r < 8; ++r) o[dt][r] *= sc[r];
    // P (C/D layout) -> LDS -> A layout; same-wave DS ops are in-order
#pragma unroll
    for (int nt = 0; nt < 8; ++nt)
#pragma unroll
      for (int r = 0; r < 8; ++r)
        Ps[w][r + koff][nt * 16 + mrow] = (bf16_t)s[nt][r];
    // O += P @ V  (B-frags direct from global Vt: contiguous key dim)
#pragma unroll
    for (int kt = 0; kt < 4; ++kt) {
      v16bf pa = make_frag(&Ps[w][mrow][kt * 32 + koff]);
#pragma unroll
      for (int dt = 0; dt < 4; ++dt) {
        const bf16_t* vr = vp + (size_t)(dt * 16 + mrow) * S_ + kb + kt * 32;
        o[dt] = wmma_bf16(pa, make_frag(vr + koff), o[dt]);
      }
    }
  }
  // finalize: role = o/l, plus rsqrt(max(||role||^2, 1e-12)) per row
  float ss[8];
#pragma unroll
  for (int r = 0; r < 8; ++r) ss[r] = 0.f;
#pragma unroll
  for (int dt = 0; dt < 4; ++dt)
#pragma unroll
    for (int r = 0; r < 8; ++r) {
      float val = o[dt][r] / lr[r];
      int srow = q0 + r + koff;
      int dk = dt * 16 + mrow;
      roleb[((size_t)bh * S_ + srow) * DK_ + dk] = (bf16_t)val;
      ss[r] += val * val;
    }
#pragma unroll
  for (int r = 0; r < 8; ++r) {
#pragma unroll
    for (int off = 1; off < 16; off <<= 1) ss[r] += __shfl_xor(ss[r], off, 32);
    if (mrow == 0)
      rsq[(size_t)bh * S_ + q0 + r + koff] = rsqrtf(fmaxf(ss[r], 1e-12f));
  }
}

// ---------------------------------------------------------------------------
// VQ: sim = role_bf16 @ cbn^T via WMMA, running argmax over V=2048,
// ws = max * rsqrt-norm, quantized row = F[id]  (gathered).
// WG = 128 threads (4 waves), 64 rows per WG.
// ---------------------------------------------------------------------------
__global__ __launch_bounds__(128) void vq_kernel(const bf16_t* __restrict__ roleb,
                                                 const bf16_t* __restrict__ cbn,
                                                 const float* __restrict__ rsq,
                                                 const float* __restrict__ Ftab,
                                                 float* __restrict__ quant,
                                                 int* __restrict__ ids,
                                                 float* __restrict__ ws) {
  __shared__ int   sIds[64];
  __shared__ float sWs[64];
  int lane = threadIdx.x & 31, w = threadIdx.x >> 5;
  int mrow = lane & 15, koff = (lane >> 4) << 3;
  int r0 = blockIdx.x * 64 + w * 16;

  v16bf a0 = make_frag(roleb + (size_t)(r0 + mrow) * DK_ + koff);
  v16bf a1 = make_frag(roleb + (size_t)(r0 + mrow) * DK_ + 32 + koff);

  float best[8]; int bidx[8];
#pragma unroll
  for (int r = 0; r < 8; ++r) { best[r] = -3.4e38f; bidx[r] = 0; }

  for (int vt = 0; vt < V_ / 16; ++vt) {
    const bf16_t* cr = cbn + (size_t)(vt * 16 + mrow) * DK_;
    v8f s;
#pragma unroll
    for (int e = 0; e < 8; ++e) s[e] = 0.f;
    s = wmma_bf16(a0, make_frag(cr + koff), s);
    s = wmma_bf16(a1, make_frag(cr + 32 + koff), s);
    int code = vt * 16 + mrow;
#pragma unroll
    for (int r = 0; r < 8; ++r)
      if (s[r] > best[r]) { best[r] = s[r]; bidx[r] = code; }  // strict > keeps first
  }
  // cross-lane argmax over the 16-lane half, tie -> lower index (jnp.argmax)
#pragma unroll
  for (int r = 0; r < 8; ++r) {
#pragma unroll
    for (int off = 1; off < 16; off <<= 1) {
      float ob = __shfl_xor(best[r], off, 32);
      int   oi = __shfl_xor(bidx[r], off, 32);
      if (ob > best[r] || (ob == best[r] && oi < bidx[r])) { best[r] = ob; bidx[r] = oi; }
    }
  }
  if (mrow == 0) {
#pragma unroll
    for (int r = 0; r < 8; ++r) {
      int rl = w * 16 + r + koff;
      sIds[rl] = bidx[r];
      sWs[rl]  = best[r] * rsq[r0 + r + koff];
    }
  }
  __syncthreads();
  int t = threadIdx.x;
  if (t < 64) {
    int gr = blockIdx.x * 64 + t;
    ids[gr] = sIds[t];
    ws[gr]  = sWs[t];
  }
  // gather quantized = F[id] : thread -> (row = t/2, 32 floats)
  int row = t >> 1, half = t & 1;
  const float4* fp = (const float4*)(Ftab + (size_t)sIds[row] * DK_ + half * 32);
  float4* qo = (float4*)(quant + ((size_t)blockIdx.x * 64 + row) * DK_ + half * 32);
#pragma unroll
  for (int j = 0; j < 8; ++j) qo[j] = fp[j];
}

// ---------------------------------------------------------------------------
extern "C" void kernel_launch(void* const* d_in, const int* in_sizes, int n_in,
                              void* d_out, int out_size, void* d_ws, size_t ws_size,
                              hipStream_t stream) {
  (void)in_sizes; (void)n_in; (void)out_size; (void)ws_size;
  const float* query  = (const float*)d_in[0];
  const float* key    = (const float*)d_in[1];
  const float* value  = (const float*)d_in[2];
  const float* mask   = (const float*)d_in[3];
  const float* q_pos  = (const float*)d_in[4];
  const float* kv_pos = (const float*)d_in[5];
  const float* Wq     = (const float*)d_in[6];
  const float* Wk     = (const float*)d_in[7];
  const float* Wv     = (const float*)d_in[8];
  const float* cb     = (const float*)d_in[9];
  const float* vqg    = (const float*)d_in[10];
  const float* vqb    = (const float*)d_in[11];
  const float* Wo     = (const float*)d_in[12];
  const float* lng    = (const float*)d_in[13];
  const float* lnb    = (const float*)d_in[14];

  const size_t BSD  = (size_t)B_ * S_ * 1024;   // 4,194,304
  const size_t DD   = (size_t)1024 * 1024;
  const size_t BHS  = (size_t)B_ * H_ * S_;     // 65,536
  const size_t BHSD = BHS * DK_;                // 4,194,304

  char* p = (char*)d_ws;
  auto take = [&](size_t bytes) { char* r = p; p += (bytes + 255) & ~(size_t)255; return r; };
  bf16_t* xq   = (bf16_t*)take(BSD * 2);
  bf16_t* xk   = (bf16_t*)take(BSD * 2);
  bf16_t* xv   = (bf16_t*)take(BSD * 2);
  bf16_t* wqb  = (bf16_t*)take(DD * 2);
  bf16_t* wkb  = (bf16_t*)take(DD * 2);
  bf16_t* wvb  = (bf16_t*)take(DD * 2);
  bf16_t* qh   = (bf16_t*)take(BHSD * 2);
  bf16_t* kh   = (bf16_t*)take(BHSD * 2);
  bf16_t* vth  = (bf16_t*)take(BHSD * 2);
  bf16_t* roleb= (bf16_t*)take(BHSD * 2);
  float*  rsq  = (float*) take(BHS * 4);
  bf16_t* cbn  = (bf16_t*)take((size_t)V_ * DK_ * 2);
  float*  Ftab = (float*) take((size_t)V_ * DK_ * 4);

  addcvt_kernel<<<(int)(BSD / 256), 256, 0, stream>>>(query, q_pos, xq, (int)BSD);
  addcvt_kernel<<<(int)(BSD / 256), 256, 0, stream>>>(key,   kv_pos, xk, (int)BSD);
  addcvt_kernel<<<(int)(BSD / 256), 256, 0, stream>>>(value, kv_pos, xv, (int)BSD);
  cvt_kernel<<<(int)(DD / 256), 256, 0, stream>>>(Wq, wqb, (int)DD);
  cvt_kernel<<<(int)(DD / 256), 256, 0, stream>>>(Wk, wkb, (int)DD);
  cvt_kernel<<<(int)(DD / 256), 256, 0, stream>>>(Wv, wvb, (int)DD);
  cb_prep<<<V_, 64, 0, stream>>>(cb, vqg, vqb, Wo, lng, lnb, cbn, Ftab);

  dim3 pg(32, 8), pb(256);
  proj_gemm<<<pg, pb, 0, stream>>>(xq, wqb, qh,  B_ * S_, 1024, 1024, S_, 0);
  proj_gemm<<<pg, pb, 0, stream>>>(xk, wkb, kh,  B_ * S_, 1024, 1024, S_, 0);
  proj_gemm<<<pg, pb, 0, stream>>>(xv, wvb, vth, B_ * S_, 1024, 1024, S_, 1);

  attn_kernel<<<dim3(S_ / 64, B_ * H_), 128, 0, stream>>>(qh, kh, vth, mask, roleb, rsq);

  float* quant = (float*)d_out;
  int*   idso  = (int*)(quant + BHSD);
  float* wso   = (float*)(quant + BHSD + BHS);
  vq_kernel<<<(int)(BHS / 64), 128, 0, stream>>>(roleb, cbn, rsq, Ftab, quant, idso, wso);
}
